// LSTM_88184268521835
// MI455X (gfx1250) — matrix operand order repaired and checked by reading, not compiled, not run on hardware
//
#include <hip/hip_runtime.h>
#include <hip/hip_bf16.h>
#include <math.h>
#include <stdint.h>

typedef __attribute__((ext_vector_type(16))) __bf16 v16bf;
typedef __attribute__((ext_vector_type(8)))  float  v8f;

#define Hh   2048
#define Ff   2048
#define Tt   4096
#define G4   8192        // 4*H
#define KD   2048        // reduction dim of GEMM (x-part) and GEMV (h-part)

static __device__ __forceinline__ uint32_t f2bf_bits(float f) {
    __bf16 b = (__bf16)f;               // RNE convert
    uint16_t u;
    __builtin_memcpy(&u, &b, 2);
    return (uint32_t)u;
}

// ---------------------------------------------------------------------------
// Pack: X -> bf16, Wx (x-columns) -> bf16, Wh (h-columns) -> swizzled bf16,
// b4, zero h/c.
// Whb swizzle: per row, memory dword p (0..1023): it=p>>7, l=(p&127)>>2,
// d=p&3, s=it*128+d*32+l; dword = bf16(W[s]) | bf16(W[s+1024])<<16.
// This makes every hs[] access in the step kernel lane-consecutive.
// ---------------------------------------------------------------------------
__global__ __launch_bounds__(256) void pack_kernel(
    const float* __restrict__ X,
    const float* __restrict__ wf, const float* __restrict__ wi,
    const float* __restrict__ wc, const float* __restrict__ wo,
    const float* __restrict__ bf, const float* __restrict__ bi,
    const float* __restrict__ bc, const float* __restrict__ bo,
    __bf16* __restrict__ Xb, __bf16* __restrict__ Wxb,
    uint32_t* __restrict__ Whb32,
    float* __restrict__ b4, float* __restrict__ h, float* __restrict__ c)
{
    size_t i      = (size_t)blockIdx.x * blockDim.x + threadIdx.x;
    size_t stride = (size_t)gridDim.x * blockDim.x;
    const size_t NX  = (size_t)Tt * Ff;
    const size_t NW  = (size_t)G4 * KD;
    const size_t NWH = (size_t)G4 * (KD / 2);    // dwords
    const float* wg[4] = {wf, wi, wc, wo};
    const float* bg[4] = {bf, bi, bc, bo};

    for (size_t p = i; p < NX; p += stride) Xb[p] = (__bf16)X[p];

    for (size_t p = i; p < NW; p += stride) {
        size_t j = p >> 11;           // row in [0,8192)
        size_t k = p & 2047;          // col in [0,2048)
        int    g = (int)(j >> 11);    // gate
        size_t r = j & 2047;          // row within gate
        Wxb[p] = (__bf16)wg[g][r * (size_t)(Hh + Ff) + Hh + k];  // x-part
    }

    for (size_t p = i; p < NWH; p += stride) {
        size_t row = p >> 10;         // 0..8191
        int    pp  = (int)(p & 1023);
        int    it  = pp >> 7;
        int    rem = pp & 127;
        int    l   = rem >> 2;
        int    d   = rem & 3;
        int    s   = it * 128 + d * 32 + l;
        int    g   = (int)(row >> 11);
        size_t r   = row & 2047;
        const float* base = wg[g] + r * (size_t)(Hh + Ff);       // h-part cols
        Whb32[p] = f2bf_bits(base[s]) | (f2bf_bits(base[s + 1024]) << 16);
    }

    for (size_t p = i; p < G4; p += stride) b4[p] = bg[p >> 11][p & 2047];
    for (size_t p = i; p < Hh; p += stride) { h[p] = 0.f; c[p] = 0.f; }
}

// ---------------------------------------------------------------------------
// Zx[t,j] = sum_k Xb[t,k] * Wxb[j,k] + b4[j]
// 128x128 tile, 256 threads (8 waves), LDS-staged, bf16 WMMA, fp32 acc.
// ---------------------------------------------------------------------------
__global__ __launch_bounds__(256) void gemm_zx_kernel(
    const __bf16* __restrict__ Xb, const __bf16* __restrict__ Wxb,
    const float* __restrict__ b4, float* __restrict__ Zx)
{
    __shared__ __align__(16) __bf16 As[128][40];   // 32 K + 8 pad (bank spread)
    __shared__ __align__(16) __bf16 Bs[128][40];

    const int tid  = threadIdx.x;
    const int wave = tid >> 5;
    const int lane = tid & 31;
    const int half = lane >> 4;
    const int hr   = lane & 15;

    const int bm = blockIdx.y * 128;   // t-tile base
    const int bn = blockIdx.x * 128;   // j-tile base

    v8f acc[8] = {};

    for (int k0 = 0; k0 < KD; k0 += 32) {
        #pragma unroll
        for (int cc = 0; cc < 2; ++cc) {
            int idx = tid + cc * 256;       // 0..511
            int row = idx >> 2;             // 0..127
            int kc  = (idx & 3) * 8;        // element offset 0,8,16,24
            *(uint4*)&As[row][kc] = *(const uint4*)&Xb [(size_t)(bm + row) * KD + k0 + kc];
            *(uint4*)&Bs[row][kc] = *(const uint4*)&Wxb[(size_t)(bn + row) * KD + k0 + kc];
        }
        __syncthreads();

        union { v16bf v; uint4 q[2]; } af;
        const int arow = wave * 16 + hr;
        af.q[0] = *(const uint4*)&As[arow][half * 8];
        af.q[1] = *(const uint4*)&As[arow][16 + half * 8];

        #pragma unroll
        for (int nb = 0; nb < 8; ++nb) {
            union { v16bf v; uint4 q[2]; } bfr;
            const int bcol = nb * 16 + hr;
            bfr.q[0] = *(const uint4*)&Bs[bcol][half * 8];
            bfr.q[1] = *(const uint4*)&Bs[bcol][16 + half * 8];
            acc[nb] = __builtin_amdgcn_wmma_f32_16x16x32_bf16(
                false, af.v, false, bfr.v, (short)0, acc[nb], false, false);
        }
        __syncthreads();
    }

    const int n_l   = lane & 15;
    const int mbase = bm + wave * 16 + 8 * half;
    #pragma unroll
    for (int nb = 0; nb < 8; ++nb) {
        const int col  = bn + nb * 16 + n_l;
        const float bb = b4[col];
        #pragma unroll
        for (int r = 0; r < 8; ++r)
            Zx[(size_t)(mbase + r) * G4 + col] = acc[nb][r] + bb;
    }
}

// ---------------------------------------------------------------------------
// One recurrence step: z_j = Zx[t,j] + dot(Whb_row_j, h); gates; update h,c.
// Block handles 16 hidden indices -> 64 gate rows; 8 waves x 8 bf16 dots each.
// Whb rows: 256 uint4 per row, k-swizzled (see pack) => conflict-free hs reads.
// ---------------------------------------------------------------------------
__global__ __launch_bounds__(256) void lstm_step_kernel(
    const uint32_t* __restrict__ Whb32,
    const float* __restrict__ Zx_t,   // Zx + t*8192
    float* __restrict__ h, float* __restrict__ c)
{
    __shared__ float hs[Hh];
    __shared__ float zs[64];

    const int tid  = threadIdx.x;
    const int wave = tid >> 5;
    const int lane = tid & 31;
    const int m0   = blockIdx.x * 16;

    for (int k = tid; k < Hh; k += 256) hs[k] = h[k];
    __syncthreads();

    #pragma unroll
    for (int s = 0; s < 8; ++s) {
        const int rr = s * 8 + wave;      // 0..63
        const int g  = rr >> 4;           // gate
        const int ml = rr & 15;           // local hidden idx
        const int j  = g * Hh + m0 + ml;  // global gate row
        const uint4* wr = (const uint4*)Whb32 + (size_t)j * 256;
        float sum = 0.f;
        #pragma unroll
        for (int it = 0; it < 8; ++it) {
            const uint4 u = wr[it * 32 + lane];
            const int   b0 = it * 128 + lane;   // s-base for dword 0
            sum += __uint_as_float(u.x << 16)          * hs[b0 +    0];
            sum += __uint_as_float(u.x & 0xffff0000u)  * hs[b0 + 1024];
            sum += __uint_as_float(u.y << 16)          * hs[b0 +   32];
            sum += __uint_as_float(u.y & 0xffff0000u)  * hs[b0 + 1056];
            sum += __uint_as_float(u.z << 16)          * hs[b0 +   64];
            sum += __uint_as_float(u.z & 0xffff0000u)  * hs[b0 + 1088];
            sum += __uint_as_float(u.w << 16)          * hs[b0 +   96];
            sum += __uint_as_float(u.w & 0xffff0000u)  * hs[b0 + 1120];
        }
        #pragma unroll
        for (int off = 16; off; off >>= 1) sum += __shfl_xor(sum, off, 32);
        if (lane == 0) zs[rr] = sum + Zx_t[j];
    }
    __syncthreads();

    if (tid < 16) {
        const int m = m0 + tid;
        const float zf = zs[0 * 16 + tid];
        const float zi = zs[1 * 16 + tid];
        const float zc = zs[2 * 16 + tid];
        const float zo = zs[3 * 16 + tid];
        const float f = 1.f / (1.f + __expf(-zf));
        const float i = 1.f / (1.f + __expf(-zi));
        const float g = tanhf(zc);
        const float o = 1.f / (1.f + __expf(-zo));
        const float cn = f * c[m] + i * g;
        c[m] = cn;
        h[m] = o * tanhf(cn);
    }
}

// ---------------------------------------------------------------------------
// y = wy @ h_T + by  (2048 x 2048 GEMV)
// ---------------------------------------------------------------------------
__global__ __launch_bounds__(256) void proj_kernel(
    const float* __restrict__ wy, const float* __restrict__ by,
    const float* __restrict__ h, float* __restrict__ y)
{
    const int tid  = threadIdx.x;
    const int wave = tid >> 5;
    const int lane = tid & 31;
    const int row  = blockIdx.x * 8 + wave;
    const float* wrow = wy + (size_t)row * Hh;
    float sum = 0.f;
    #pragma unroll 8
    for (int k = lane; k < Hh; k += 32) sum += wrow[k] * h[k];
    #pragma unroll
    for (int off = 16; off; off >>= 1) sum += __shfl_xor(sum, off, 32);
    if (lane == 0) y[row] = by[row] + sum;
}

// ---------------------------------------------------------------------------
extern "C" void kernel_launch(void* const* d_in, const int* in_sizes, int n_in,
                              void* d_out, int out_size, void* d_ws, size_t ws_size,
                              hipStream_t stream) {
    // inputs: X, wf, bf, wi, bi, wc, bc, wo, bo, wy, by   (all fp32)
    const float* X  = (const float*)d_in[0];
    const float* wf = (const float*)d_in[1];
    const float* bf = (const float*)d_in[2];
    const float* wi = (const float*)d_in[3];
    const float* bi = (const float*)d_in[4];
    const float* wc = (const float*)d_in[5];
    const float* bc = (const float*)d_in[6];
    const float* wo = (const float*)d_in[7];
    const float* bo = (const float*)d_in[8];
    const float* wy = (const float*)d_in[9];
    const float* by = (const float*)d_in[10];
    float* y = (float*)d_out;

    // workspace layout
    char* ws = (char*)d_ws;
    size_t off = 0;
    float*    Zx    = (float*)(ws + off);    off += (size_t)Tt * G4 * 4;   // 128 MB
    __bf16*   Xb    = (__bf16*)(ws + off);   off += (size_t)Tt * Ff * 2;   //  16 MB
    __bf16*   Wxb   = (__bf16*)(ws + off);   off += (size_t)G4 * KD * 2;   //  32 MB
    uint32_t* Whb32 = (uint32_t*)(ws + off); off += (size_t)G4 * (KD/2)*4; //  32 MB
    float*    b4    = (float*)(ws + off);    off += (size_t)G4 * 4;        //  32 KB
    float*    h     = (float*)(ws + off);    off += (size_t)Hh * 4;        //   8 KB
    float*    c     = (float*)(ws + off);                                  //   8 KB

    // 1) pack bf16 operands, bias, zero state
    pack_kernel<<<4096, 256, 0, stream>>>(X, wf, wi, wc, wo, bf, bi, bc, bo,
                                          Xb, Wxb, Whb32, b4, h, c);

    // 2) Zx = X @ Wx^T + b4   (bf16 WMMA, fp32 accumulate)
    dim3 ggrid(G4 / 128, Tt / 128);   // (64, 32)
    gemm_zx_kernel<<<ggrid, 256, 0, stream>>>(Xb, Wxb, b4, Zx);

    // 3) sequential recurrence: 4096 step launches (graph-captured once)
    for (int t = 0; t < Tt; ++t) {
        lstm_step_kernel<<<Hh / 16, 256, 0, stream>>>(
            Whb32, Zx + (size_t)t * G4, h, c);
    }

    // 4) final projection
    proj_kernel<<<Hh / 8, 256, 0, stream>>>(wy, by, h, y);
}
